// MultiHeadAttention_45921790329378
// MI455X (gfx1250) — matrix operand order repaired and checked by reading, not compiled, not run on hardware
//
#include <hip/hip_runtime.h>
#include <hip/hip_bf16.h>

// ---------------------------------------------------------------------------
// MultiHeadAttention for MI455X (gfx1250), wave32 + WMMA bf16.
// Pipeline: wconv -> layernorm(x3) -> proj GEMM(x3, Q pre-scaled by 1/8)
//           -> V-transpose
//           -> fused [scores -> softmax -> attn@V] (256KB LDS per block,
//              double-buffered fragment loads)
//           -> fc GEMM(x2)
// ---------------------------------------------------------------------------

typedef __bf16 bf16;
typedef __bf16 bf16x2  __attribute__((ext_vector_type(2)));
typedef __bf16 bf16x8  __attribute__((ext_vector_type(8)));
typedef __bf16 bf16x16 __attribute__((ext_vector_type(16)));
typedef float  f32x2   __attribute__((ext_vector_type(2)));
typedef float  v8f     __attribute__((ext_vector_type(8)));

#define NHEAD 8
#define DHEAD 64
#define BATCH 8
#define SEQ   1024
#define DIN   512
#define MROWS (BATCH * SEQ)   // 8192

// Fused-attention LDS: 64 rows; each row owns 4096 bytes.
//   phase 1/2: fp32 scores  row r -> bytes [4096r, 4096r+4096)
//   phase 2/3: bf16 probs   row r -> bytes [4096r, 4096r+2048)   (same slot)
#define SROW_BYTES 4096
#define FUSED_LDS  (64 * SROW_BYTES)   // 256 KB  (WGP allows 320 KB)

// -------------------------- WMMA helpers -----------------------------------

static __device__ __forceinline__ v8f wmma_bf16(bf16x16 a, bf16x16 b, v8f c) {
  // 8 args: (neg_a, A, neg_b, B, c_mod, C, reuse_a, reuse_b)
  return __builtin_amdgcn_wmma_f32_16x16x32_bf16(false, a, false, b,
                                                 (short)0, c, false, false);
}

// 16-bit A-matrix 16x32 layout: lane L holds row M=L%16,
// K = half*16 + (0..15) contiguous (matches B fragments).
// Caller passes base already offset by half*16.
static __device__ __forceinline__ bf16x16 load_a_frag(const bf16* base) {
  bf16x8 lo = *(const bf16x8*)(base);
  bf16x8 hi = *(const bf16x8*)(base + 8);
  bf16x16 r;
#pragma unroll
  for (int i = 0; i < 8; ++i) { r[i] = lo[i]; r[i + 8] = hi[i]; }
  return r;
}

// B-matrix 32x16: lane L<16 holds col N=L with K=0..15 contiguous;
// lanes 16..31 hold K=16..31.  Caller passes base offset by half*16.
static __device__ __forceinline__ bf16x16 load_b_frag(const bf16* base) {
  return *(const bf16x16*)(base);
}

// -------------------------- small kernels ----------------------------------

// fp32 W[k][n] (512x512) -> bf16 Wt[n][k]
__global__ __launch_bounds__(256) void wconv_kernel(const float* __restrict__ W,
                                                    bf16* __restrict__ Wt) {
  int idx = blockIdx.x * 256 + threadIdx.x;   // idx = n*512 + k
  int n = idx >> 9, kk = idx & 511;
  Wt[idx] = (bf16)W[kk * 512 + n];
}

// LayerNorm over rows of 512, fp32 in -> bf16 out.  One 256-thread block/row.
__global__ __launch_bounds__(256) void ln_kernel(const float* __restrict__ x,
                                                 const float* __restrict__ g,
                                                 const float* __restrict__ bia,
                                                 bf16* __restrict__ out) {
  int row = blockIdx.x;
  int t = threadIdx.x;
  const float* xr = x + (size_t)row * DIN;
  float v0 = xr[t], v1 = xr[t + 256];
  float s = v0 + v1;
  float sq = v0 * v0 + v1 * v1;
#pragma unroll
  for (int off = 16; off > 0; off >>= 1) {
    s  += __shfl_xor(s, off, 32);
    sq += __shfl_xor(sq, off, 32);
  }
  __shared__ float red[2][8];
  int w = t >> 5, lane = t & 31;
  if (lane == 0) { red[0][w] = s; red[1][w] = sq; }
  __syncthreads();
  if (t == 0) {
    float ts = 0.f, tq = 0.f;
#pragma unroll
    for (int i = 0; i < 8; ++i) { ts += red[0][i]; tq += red[1][i]; }
    red[0][0] = ts; red[1][0] = tq;
  }
  __syncthreads();
  float mean = red[0][0] * (1.0f / DIN);
  float var  = red[1][0] * (1.0f / DIN) - mean * mean;
  float rstd = rsqrtf(var + 1e-5f);
  bf16* orow = out + (size_t)row * DIN;
  orow[t]       = (bf16)((v0 - mean) * rstd * g[t] + bia[t]);
  orow[t + 256] = (bf16)((v1 - mean) * rstd * g[t + 256] + bia[t + 256]);
}

// Vp row-major [b*L+lk][h*64+d] -> Vt [hb][d][lk] (contiguous B frags).
__global__ __launch_bounds__(256) void vtrans_kernel(const bf16* __restrict__ Vp,
                                                     bf16* __restrict__ Vt) {
  int tid = blockIdx.x * 256 + threadIdx.x;       // (hb, d, lk), lk fastest
  int hb = tid >> 16;
  int d  = (tid >> 10) & 63;
  int lk = tid & (SEQ - 1);
  int h = hb >> 3, bb = hb & 7;
  Vt[tid] = Vp[(size_t)(bb * SEQ + lk) * 512 + h * DHEAD + d];
}

// -------------------------- GEMM (templated epilogue) ----------------------
// C[M x 512] = A[M x 512] * Wt^T.  A bf16 row-major, Wt bf16 [n][k].
// Block tile 64(M) x 128(N); 8 waves each own a 32x32 subtile (2x2 frags).
// PRESCALE multiplies the result by 0.125 (exact power of two) so the
// softmax scale never touches the fused attention hot loop.
template <bool WRITE_F32, bool PRESCALE>
__global__ __launch_bounds__(256) void gemm_kernel(const bf16* __restrict__ A,
                                                   const bf16* __restrict__ Bt,
                                                   float* __restrict__ Cf,
                                                   bf16* __restrict__ Cb) {
  const int lda = 512, ldb = 512, ldc = 512;
  int tid = threadIdx.x;
  int w = tid >> 5, lane = tid & 31;
  int half = lane >> 4, l16 = lane & 15;
  int wm = w >> 2, wn = w & 3;
  int m0 = blockIdx.x * 64 + wm * 32;
  int n0 = blockIdx.y * 128 + wn * 32;

  const bf16* a0p = A  + (size_t)(m0 + l16) * lda + half * 16;
  const bf16* a1p = a0p + (size_t)16 * lda;
  const bf16* b0p = Bt + (size_t)(n0 + l16) * ldb + half * 16;
  const bf16* b1p = b0p + (size_t)16 * ldb;

  v8f acc[2][2] = {};
#pragma unroll 2
  for (int k0 = 0; k0 < 512; k0 += 32) {
    if (k0 + 32 < 512) {
      __builtin_prefetch(a0p + k0 + 32, 0, 1);
      __builtin_prefetch(b0p + k0 + 32, 0, 1);
    }
    bf16x16 a0 = load_a_frag(a0p + k0);
    bf16x16 a1 = load_a_frag(a1p + k0);
    bf16x16 b0 = load_b_frag(b0p + k0);
    bf16x16 b1 = load_b_frag(b1p + k0);
    acc[0][0] = wmma_bf16(a0, b0, acc[0][0]);
    acc[0][1] = wmma_bf16(a0, b1, acc[0][1]);
    acc[1][0] = wmma_bf16(a1, b0, acc[1][0]);
    acc[1][1] = wmma_bf16(a1, b1, acc[1][1]);
  }
#pragma unroll
  for (int mi = 0; mi < 2; ++mi)
#pragma unroll
    for (int nj = 0; nj < 2; ++nj) {
      int col = n0 + nj * 16 + l16;
#pragma unroll
      for (int vv = 0; vv < 8; ++vv) {
        int row = m0 + mi * 16 + vv + half * 8;
        float val = acc[mi][nj][vv];
        if constexpr (PRESCALE) val *= 0.125f;
        if constexpr (WRITE_F32)
          Cf[(size_t)row * ldc + col] = val;
        else
          Cb[(size_t)row * ldc + col] = (bf16)val;
      }
    }
}

// -------------------------- fused attention --------------------------------
// One block per (hb, 64 query rows), 8 waves, 256KB dynamic LDS.
// Phase 1: S = (Q/8).K -> LDS fp32 (scale pre-folded into Qp);
//          B-fragments double-buffered so loads overlap WMMA + DS stores.
// Phase 2: diag mask + softmax; fp32 probs -> global attn (mandatory output),
//          bf16 probs compacted into each row's own LDS slot.
// Phase 3: O = P @ V with A-frags from LDS (ds_load), B-frags from Vt;
//          both double-buffered.
__global__ __launch_bounds__(256) void fused_attn_kernel(
    const bf16* __restrict__ Qp, const bf16* __restrict__ Kp,
    const bf16* __restrict__ Vt, float* __restrict__ attn,
    bf16* __restrict__ O) {
  extern __shared__ char smem[];
  int hb = blockIdx.y;
  int h = hb >> 3, bb = hb & 7;
  int tid = threadIdx.x;
  int w = tid >> 5, lane = tid & 31;
  int half = lane >> 4, l16 = lane & 15;
  int lq_blk = blockIdx.x * 64;

  // ---- Phase 1: scores into LDS --------------------------------------
  {
    int mi = w & 3, ch = w >> 2;             // row tile, column half
    int r0 = mi * 16;                        // local row base
    const bf16* aQ = Qp + (size_t)(bb * SEQ + lq_blk + r0 + l16) * 512
                        + h * DHEAD + half * 16;
    bf16x16 a0 = load_a_frag(aQ);            // d in [0,32)
    bf16x16 a1 = load_a_frag(aQ + 32);       // d in [32,64)

    const bf16* Kbase = Kp + ((size_t)bb * SEQ + l16) * 512
                           + h * DHEAD + half * 16;
    // tile t covers K-rows [(ch*32+t)*16, +16)
    auto kptr = [&](int t) {
      return Kbase + (size_t)((ch * 32 + t) * 16) * 512;
    };
    bf16x16 b0 = load_b_frag(kptr(0));
    bf16x16 b1 = load_b_frag(kptr(0) + 32);
    for (int t = 0; t < 32; ++t) {
      int tn = (t + 1 < 32) ? t + 1 : t;     // branch-free tail
      bf16x16 nb0 = load_b_frag(kptr(tn));
      bf16x16 nb1 = load_b_frag(kptr(tn) + 32);
      v8f acc = {};
      acc = wmma_bf16(a0, b0, acc);
      acc = wmma_bf16(a1, b1, acc);
      int col = (ch * 32 + t) * 16 + l16;
#pragma unroll
      for (int vv = 0; vv < 8; ++vv) {
        int r = r0 + vv + half * 8;
        *(float*)(smem + r * SROW_BYTES + col * 4) = acc[vv];
      }
      b0 = nb0; b1 = nb1;
    }
  }
  __syncthreads();

  // ---- Phase 2: softmax (wave per row), write attn + bf16 probs -------
  for (int r = w; r < 64; r += 8) {
    int qglob = lq_blk + r;                  // global query row == diag col
    const float* srow = (const float*)(smem + r * SROW_BYTES);
    float p[32];
    float mx = -3.4e38f;
#pragma unroll
    for (int i = 0; i < 16; ++i) {
      int c = 2 * lane + 64 * i;
      float v0 = srow[c], v1 = srow[c + 1];
      if (c == qglob)     v0 = -1e32f;
      if (c + 1 == qglob) v1 = -1e32f;
      p[2 * i] = v0; p[2 * i + 1] = v1;
      mx = fmaxf(mx, fmaxf(v0, v1));
    }
#pragma unroll
    for (int off = 16; off > 0; off >>= 1) mx = fmaxf(mx, __shfl_xor(mx, off, 32));
    float sum = 0.f;
#pragma unroll
    for (int i = 0; i < 32; ++i) { p[i] = __expf(p[i] - mx); sum += p[i]; }
#pragma unroll
    for (int off = 16; off > 0; off >>= 1) sum += __shfl_xor(sum, off, 32);
    float inv = 1.0f / sum;
    float* grow = attn + ((size_t)hb * SEQ + qglob) * SEQ;
    bf16x2* brow = (bf16x2*)(smem + r * SROW_BYTES);   // bf16 slot of SAME row
#pragma unroll
    for (int i = 0; i < 16; ++i) {
      int c = 2 * lane + 64 * i;
      float v0 = p[2 * i] * inv, v1 = p[2 * i + 1] * inv;
      f32x2 fg = {v0, v1};
      *(f32x2*)(grow + c) = fg;              // coalesced 8B stores
      bf16x2 bp = {(bf16)v0, (bf16)v1};
      brow[c >> 1] = bp;                     // compacted in-row (no aliasing)
    }
  }
  __syncthreads();

  // ---- Phase 3: O = P @ V, A-frags from LDS ---------------------------
  {
    int mi = w & 3, nh = w >> 2;
    int r0 = mi * 16;
    int n0 = nh * 32;
    const bf16* arow = (const bf16*)(smem + (r0 + l16) * SROW_BYTES) + half * 16;
    const bf16* bbase = Vt + (size_t)hb * (DHEAD * SEQ)
                           + (size_t)(n0 + l16) * SEQ + half * 16;
    v8f acc[2] = {};
    bf16x16 a  = load_a_frag(arow);
    bf16x16 b0 = load_b_frag(bbase);
    bf16x16 b1 = load_b_frag(bbase + (size_t)16 * SEQ);
    for (int k0 = 0; k0 < SEQ; k0 += 32) {
      int kn = (k0 + 32 < SEQ) ? k0 + 32 : k0;   // branch-free tail
      bf16x16 na  = load_a_frag(arow + kn);
      bf16x16 nb0 = load_b_frag(bbase + kn);
      bf16x16 nb1 = load_b_frag(bbase + (size_t)16 * SEQ + kn);
      acc[0] = wmma_bf16(a, b0, acc[0]);
      acc[1] = wmma_bf16(a, b1, acc[1]);
      a = na; b0 = nb0; b1 = nb1;
    }
#pragma unroll
    for (int nj = 0; nj < 2; ++nj) {
      int d = n0 + nj * 16 + l16;
#pragma unroll
      for (int vv = 0; vv < 8; ++vv) {
        int lq = lq_blk + r0 + vv + half * 8;
        O[(size_t)(bb * SEQ + lq) * 512 + h * DHEAD + d] = (bf16)acc[nj][vv];
      }
    }
  }
}

// -------------------------- launch -----------------------------------------

extern "C" void kernel_launch(void* const* d_in, const int* in_sizes, int n_in,
                              void* d_out, int out_size, void* d_ws, size_t ws_size,
                              hipStream_t stream) {
  (void)in_sizes; (void)n_in; (void)out_size; (void)ws_size;
  const float* q    = (const float*)d_in[0];
  const float* k    = (const float*)d_in[1];
  const float* v    = (const float*)d_in[2];
  const float* w_q  = (const float*)d_in[3];
  const float* w_k  = (const float*)d_in[4];
  const float* w_v  = (const float*)d_in[5];
  const float* w_f1 = (const float*)d_in[6];
  const float* w_f2 = (const float*)d_in[7];
  const float* ln1g = (const float*)d_in[8];
  const float* ln1b = (const float*)d_in[9];
  const float* ln2g = (const float*)d_in[10];
  const float* ln2b = (const float*)d_in[11];
  const float* ln3g = (const float*)d_in[12];
  const float* ln3b = (const float*)d_in[13];

  // workspace carve-up (bf16 buffers), 256B aligned
  char* ws = (char*)d_ws;
  size_t off = 0;
  auto carve = [&](size_t bytes) {
    void* p = ws + off;
    off = (off + bytes + 255) & ~(size_t)255;
    return p;
  };
  const size_t WBYTES = (size_t)512 * 512 * sizeof(bf16);
  const size_t MBYTES = (size_t)MROWS * 512 * sizeof(bf16);
  bf16* wt_q  = (bf16*)carve(WBYTES);
  bf16* wt_k  = (bf16*)carve(WBYTES);
  bf16* wt_v  = (bf16*)carve(WBYTES);
  bf16* wt_f1 = (bf16*)carve(WBYTES);
  bf16* wt_f2 = (bf16*)carve(WBYTES);
  bf16* qn = (bf16*)carve(MBYTES);
  bf16* kn = (bf16*)carve(MBYTES);
  bf16* vn = (bf16*)carve(MBYTES);
  bf16* Qp = (bf16*)carve(MBYTES);
  bf16* Kp = (bf16*)carve(MBYTES);
  bf16* Vp = (bf16*)carve(MBYTES);
  bf16* Vt = (bf16*)carve(MBYTES);   // 64*64*1024 == 8192*512
  bf16* AV = (bf16*)carve(MBYTES);

  float* dyn  = (float*)d_out;                                // 8192*512
  float* stat = dyn + (size_t)MROWS * 512;                    // 8192*512
  float* attn = stat + (size_t)MROWS * 512;                   // 64*1024*1024

  // 1) weights -> bf16 transposed
  wconv_kernel<<<1024, 256, 0, stream>>>(w_q,  wt_q);
  wconv_kernel<<<1024, 256, 0, stream>>>(w_k,  wt_k);
  wconv_kernel<<<1024, 256, 0, stream>>>(w_v,  wt_v);
  wconv_kernel<<<1024, 256, 0, stream>>>(w_f1, wt_f1);
  wconv_kernel<<<1024, 256, 0, stream>>>(w_f2, wt_f2);

  // 2) layernorms
  ln_kernel<<<MROWS, 256, 0, stream>>>(q, ln1g, ln1b, qn);
  ln_kernel<<<MROWS, 256, 0, stream>>>(k, ln2g, ln2b, kn);
  ln_kernel<<<MROWS, 256, 0, stream>>>(v, ln3g, ln3b, vn);

  // 3) projections (bf16 out); Q pre-scaled by 1/8 (exact in bf16)
  dim3 gg(MROWS / 64, 512 / 128);
  gemm_kernel<false, true ><<<gg, 256, 0, stream>>>(qn, wt_q, nullptr, Qp);
  gemm_kernel<false, false><<<gg, 256, 0, stream>>>(kn, wt_k, nullptr, Kp);
  gemm_kernel<false, false><<<gg, 256, 0, stream>>>(vn, wt_v, nullptr, Vp);

  // 4) V transpose (for contiguous B fragments in attn@V)
  vtrans_kernel<<<(NHEAD * BATCH * DHEAD * SEQ) / 256, 256, 0, stream>>>(Vp, Vt);

  // 5) fused scores + softmax + attn@V  (256 KB LDS per block)
  fused_attn_kernel<<<dim3(SEQ / 64, NHEAD * BATCH), 256, FUSED_LDS, stream>>>(
      Qp, Kp, Vt, attn, AV);

  // 6) output GEMMs (fp32 out)
  gemm_kernel<true, false><<<gg, 256, 0, stream>>>(AV, wt_f1, dyn,  nullptr);
  gemm_kernel<true, false><<<gg, 256, 0, stream>>>(Vp, wt_f2, stat, nullptr);
}
